// RPNModule_57466662420868
// MI455X (gfx1250) — matrix operand order repaired and checked by reading, not compiled, not run on hardware
//
#include <hip/hip_runtime.h>
#include <hip/hip_bf16.h>

// ---------------- constants ----------------
#define B_    4
#define H_    188
#define W_    188
#define P_    (H_*W_)       // 35344 spatial positions per batch
#define IC0   384
#define OC_   64
#define NCLS  3
#define K_    500
#define NCH   11            // 2+1+3+2+3 prediction channels
#define SELW  ((NCLS*P_+31)/32)

#ifndef __has_builtin
#define __has_builtin(x) 0
#endif
#if __has_builtin(__builtin_amdgcn_global_load_async_to_lds_b128)
#define USE_ASYNC_LDS 1
#else
#define USE_ASYNC_LDS 0
#endif

typedef __attribute__((ext_vector_type(16))) __bf16 v16bf;
typedef __attribute__((ext_vector_type(8)))  float  v8f;
typedef __attribute__((ext_vector_type(4)))  int    v4i;
typedef __attribute__((address_space(1))) v4i g_v4i;   // global int4
typedef __attribute__((address_space(3))) v4i l_v4i;   // LDS int4

struct Bf32B { uint4 lo, hi; };   // 32 bytes == one bf16 WMMA fragment

__device__ __forceinline__ unsigned short f2bf(float f) {
  union { float f; unsigned u; } a; a.f = f;
  unsigned u = a.u;
  return (unsigned short)((u + 0x7fffu + ((u >> 16) & 1u)) >> 16);   // RNE
}
__device__ __forceinline__ float bf2f(unsigned short h) {
  union { unsigned u; float f; } a; a.u = ((unsigned)h) << 16;
  return a.f;
}
__device__ __forceinline__ void wait_async0() {
#if __has_builtin(__builtin_amdgcn_s_wait_asynccnt)
  __builtin_amdgcn_s_wait_asynccnt(0);
#else
  asm volatile("s_wait_asynccnt 0" ::: "memory");
#endif
}

// ---------------- BN/bias folding: scale = g/sqrt(v+eps), shift = (b-m)*scale + be ----------------
__global__ void bn_coef_kernel(const float* __restrict__ g, const float* __restrict__ be,
                               const float* __restrict__ m, const float* __restrict__ v,
                               const float* __restrict__ bconv, float* __restrict__ outc) {
  int n = threadIdx.x;
  if (n < 64) {
    float a = g[n] / sqrtf(v[n] + 1e-5f);
    outc[n]      = a;
    outc[64 + n] = (bconv[n] - m[n]) * a + be[n];
  }
}

// ---------------- implicit-GEMM 3x3 conv + BN + ReLU via v_wmma_f32_16x16x32_bf16 ----------------
// M = 128 spatial positions per block (8 waves x 16 rows), N = 64 out channels, K = IC*9 in chunks of 32.
// Output layout: NHWC bf16 [B][P][64] so head convs read 32 contiguous channels per position.
template<int IC, bool IN_NCHW_F32>
__global__ __launch_bounds__(256)
void conv3x3_bn_relu_wmma(const void* __restrict__ inRaw,
                          const float* __restrict__ wgt,     // [64][IC][3][3] f32
                          const float* __restrict__ coefs,   // scale[64], shift[64]
                          unsigned short* __restrict__ out)  // [B][P][64] bf16
{
  __shared__ __align__(16) unsigned short lds_a[128 * 32];   // [pos][k]   8 KB
  __shared__ __align__(16) unsigned short lds_b[64 * 32];    // [n][k]     4 KB

  const int tile = blockIdx.x;
  const int b    = blockIdx.y;
  const int tid  = threadIdx.x;
  const int lane = tid & 31;
  const int wv   = tid >> 5;
  const int l16  = lane & 15;
  const bool hi  = lane >= 16;
  const int pbaseTile = tile * 128;

  v8f acc[4];
  #pragma unroll
  for (int nb = 0; nb < 4; ++nb)
    #pragma unroll
    for (int r = 0; r < 8; ++r) acc[nb][r] = 0.f;

  for (int tap = 0; tap < 9; ++tap) {
    const int dy = tap / 3 - 1, dx = tap % 3 - 1;
    const int ky = tap / 3,     kx = tap % 3;
    for (int cb = 0; cb < IC; cb += 32) {
      // ---- stage A: 128 positions x 32 input channels (SAME padding -> zeros) ----
      if constexpr (IN_NCHW_F32) {
        // Each thread: one position, a contiguous 16-channel span; pack 2 bf16/dword
        // -> 8x ds_store_b32 instead of 16x ds_store_b16.
        const float* in = (const float*)inRaw;
        int pos = tid & 127;
        int p = pbaseTile + pos;
        int y = p / W_, x = p % W_;
        int yy = y + dy, xx = x + dx;
        bool valid = (p < P_) && ((unsigned)yy < (unsigned)H_) && ((unsigned)xx < (unsigned)W_);
        int cBase = (tid >> 7) * 16;   // 0 or 16
        size_t base = valid ? ((size_t)((b * IC + cb + cBase) * H_ + yy) * W_ + xx) : 0;
        unsigned int* s32 = (unsigned int*)&lds_a[pos * 32 + cBase];
        #pragma unroll
        for (int j = 0; j < 8; ++j) {
          float v0 = valid ? in[base + (size_t)(2 * j) * (H_ * W_)] : 0.f;
          float v1 = valid ? in[base + (size_t)(2 * j + 1) * (H_ * W_)] : 0.f;
          s32[j] = (unsigned)f2bf(v0) | ((unsigned)f2bf(v1) << 16);
        }
      } else {
        // bf16 NHWC input: straight global->LDS copy; use CDNA5 async-LDS path if available.
        const unsigned short* xin = (const unsigned short*)inRaw;
        int pos = tid >> 1;
        int h16 = (tid & 1) * 16;
        int p = pbaseTile + pos;
        int y = p / W_, x = p % W_;
        int yy = y + dy, xx = x + dx;
        bool valid = (p < P_) && ((unsigned)yy < (unsigned)H_) && ((unsigned)xx < (unsigned)W_);
        unsigned short* sp_ = &lds_a[pos * 32 + h16];
        if (valid) {
          const unsigned short* gp =
              &xin[((size_t)(b * P_ + yy * W_ + xx)) * 64 + cb + h16];
#if USE_ASYNC_LDS
          __builtin_amdgcn_global_load_async_to_lds_b128((g_v4i*)gp,       (l_v4i*)sp_,       0, 0);
          __builtin_amdgcn_global_load_async_to_lds_b128((g_v4i*)(gp + 8), (l_v4i*)(sp_ + 8), 0, 0);
#else
          const uint4* g4 = (const uint4*)gp;
          uint4* s4 = (uint4*)sp_;
          s4[0] = g4[0]; s4[1] = g4[1];
#endif
        } else {
          uint4 z = {0, 0, 0, 0};
          uint4* s4 = (uint4*)sp_;
          s4[0] = z; s4[1] = z;
        }
      }
      // ---- stage B: weights for this tap, transposed to [n][k] bf16 ----
      {
        int o  = tid >> 2;
        int k0 = (tid & 3) * 8;
        #pragma unroll
        for (int i = 0; i < 8; i += 2) {
          int k = k0 + i;
          float v0 = wgt[((size_t)(o * IC + cb + k) * 3 + ky) * 3 + kx];
          float v1 = wgt[((size_t)(o * IC + cb + k + 1) * 3 + ky) * 3 + kx];
          *(unsigned int*)&lds_b[o * 32 + k] =
              (unsigned)f2bf(v0) | ((unsigned)f2bf(v1) << 16);
        }
      }
      if constexpr (!IN_NCHW_F32) {
#if USE_ASYNC_LDS
        wait_async0();   // ASYNCcnt==0: this wave's async LDS writes are done
#endif
      }
      __syncthreads();
      // ---- build fragments per ISA lane layouts, issue 4 WMMAs ----
      {
        int row = wv * 16 + l16;
        int kbA = hi ? 8 : 0;                       // A: K 0-7/16-23 (lanes 0-15) vs 8-15/24-31
        Bf32B au;
        au.lo = *(const uint4*)&lds_a[row * 32 + kbA];
        au.hi = *(const uint4*)&lds_a[row * 32 + kbA + 16];
        v16bf afrag = __builtin_bit_cast(v16bf, au);
        int kbB = hi ? 8 : 0;                       // B: K 0-7/16-23 (lanes 0-15) vs 8-15/24-31
        #pragma unroll
        for (int nb = 0; nb < 4; ++nb) {
          int n = nb * 16 + l16;
          Bf32B bu;
          bu.lo = *(const uint4*)&lds_b[n * 32 + kbB];
          bu.hi = *(const uint4*)&lds_b[n * 32 + kbB + 16];
          v16bf bfrag = __builtin_bit_cast(v16bf, bu);
          acc[nb] = __builtin_amdgcn_wmma_f32_16x16x32_bf16(
              false, afrag, false, bfrag, (short)0, acc[nb], false, false);
        }
      }
      __syncthreads();
    }
  }
  // ---- epilogue: fused BN scale/shift + ReLU, coalesced NHWC bf16 store ----
  int pbase = pbaseTile + wv * 16 + (hi ? 8 : 0);
  #pragma unroll
  for (int nb = 0; nb < 4; ++nb) {
    int n = nb * 16 + l16;
    float sc = coefs[n], sh = coefs[64 + n];
    #pragma unroll
    for (int r = 0; r < 8; ++r) {
      int p = pbase + r;
      if (p < P_) {
        float v = fmaxf(acc[nb][r] * sc + sh, 0.f);
        out[((size_t)(b * P_ + p)) * 64 + n] = f2bf(v);
      }
    }
  }
}

// ---------------- final per-head 3x3 conv 64 -> oc (oc <= 3): direct VALU ----------------
__global__ __launch_bounds__(256)
void head_final_conv(const unsigned short* __restrict__ hbuf,  // NHWC bf16 [B][P][64]
                     const float* __restrict__ wf,             // [oc][64][3][3]
                     const float* __restrict__ bfb,            // [oc]
                     float* __restrict__ preds, int oc, int chBase)
{
  int id = blockIdx.x * 256 + threadIdx.x;
  int total = B_ * P_ * oc;
  if (id >= total) return;
  int o = id % oc;
  int rem = id / oc;
  int p = rem % P_;
  int b = rem / P_;
  int y = p / W_, x = p % W_;
  float s = bfb[o];
  for (int ky = 0; ky < 3; ++ky) {
    int yy = y + ky - 1; if ((unsigned)yy >= (unsigned)H_) continue;
    for (int kx = 0; kx < 3; ++kx) {
      int xx = x + kx - 1; if ((unsigned)xx >= (unsigned)W_) continue;
      const unsigned short* hp = &hbuf[((size_t)(b * P_ + yy * W_ + xx)) * 64];
      const float* wp = &wf[(size_t)(o * 64) * 9 + ky * 3 + kx];
      #pragma unroll 8
      for (int c = 0; c < 64; ++c)
        s += bf2f(hp[c]) * wp[c * 9];
    }
  }
  preds[(size_t)(b * NCH + chBase + o) * P_ + p] = s;
}

// ---------------- sigmoid on heatmap channels (preds ch 8..10) ----------------
__global__ void hm_sigmoid(float* __restrict__ preds) {
  int id = blockIdx.x * 256 + threadIdx.x;
  int tot = B_ * NCLS * P_;
  if (id >= tot) return;
  int b = id / (NCLS * P_);
  int r = id % (NCLS * P_);
  size_t a = (size_t)(b * NCH + 8) * P_ + r;
  preds[a] = 1.f / (1.f + expf(-preds[a]));
}

// ---------------- exact top-500 per batch: iterative argmax with LDS selected-bitmap ----------------
__global__ __launch_bounds__(1024)
void topk_kernel(const float* __restrict__ preds, float* __restrict__ tscore, int* __restrict__ tind) {
  __shared__ float rv[1024];
  __shared__ int   ri[1024];
  __shared__ unsigned sel[SELW];
  int b = blockIdx.x;
  int tid = threadIdx.x;
  const float* sc = preds + (size_t)(b * NCH + 8) * P_;   // 3*P contiguous, matches c*H*W+sp flat
  for (int i = tid; i < SELW; i += 1024) sel[i] = 0u;
  __syncthreads();
  for (int k = 0; k < K_; ++k) {
    float bv = -1e30f; int bi = 0x7fffffff;
    for (int i = tid; i < NCLS * P_; i += 1024) {
      if ((sel[i >> 5] >> (i & 31)) & 1u) continue;
      float v = sc[i];
      if (v > bv || (v == bv && i < bi)) { bv = v; bi = i; }
    }
    rv[tid] = bv; ri[tid] = bi;
    __syncthreads();
    for (int s = 512; s > 0; s >>= 1) {
      if (tid < s) {
        float ov = rv[tid + s]; int oi = ri[tid + s];
        if (ov > rv[tid] || (ov == rv[tid] && oi < ri[tid])) { rv[tid] = ov; ri[tid] = oi; }
      }
      __syncthreads();
    }
    if (tid == 0) {
      tscore[b * K_ + k] = rv[0];
      tind[b * K_ + k]   = ri[0];
      sel[ri[0] >> 5] |= (1u << (ri[0] & 31));
    }
    __syncthreads();
  }
}

// ---------------- gather + box decode (also zeroes keepAny) ----------------
__global__ void decode_kernel(const float* __restrict__ preds,
                              const float* __restrict__ tscore, const int* __restrict__ tind,
                              float* __restrict__ boxes, float* __restrict__ bev4,
                              int* __restrict__ maskv, int* __restrict__ clsv,
                              int* __restrict__ keepAny) {
  int id = blockIdx.x * 256 + threadIdx.x;
  if (id >= B_ * K_) return;
  keepAny[id] = 0;
  int b = id / K_;
  float score = tscore[id];
  int ind = tind[id];
  int cls = ind / P_;
  int sp  = ind % P_;
  float ys = (float)(sp / W_);
  float xs = (float)(sp % W_);
  const float* pb = preds + (size_t)b * NCH * P_;
  float c0 = pb[0 * P_ + sp], c1 = pb[1 * P_ + sp];
  float cz = pb[2 * P_ + sp];
  float d0 = expf(pb[3 * P_ + sp]);
  float d1 = expf(pb[4 * P_ + sp]);
  float d2 = expf(pb[5 * P_ + sp]);
  float rc = pb[6 * P_ + sp], rs = pb[7 * P_ + sp];
  float ang = atan2f(rs, rc);
  float xw = (xs + c0) * 0.8f + (-75.2f);   // STRIDE*VOX = 8*0.1
  float yw = (ys + c1) * 0.8f + (-75.2f);
  float* bx = boxes + id * 7;
  bx[0] = xw; bx[1] = yw; bx[2] = cz; bx[3] = d0; bx[4] = d1; bx[5] = d2; bx[6] = ang;
  bool m = (xw >= -80.f) && (yw >= -80.f) && (cz >= -10.f) &&
           (xw <=  80.f) && (yw <=  80.f) && (cz <=  10.f) && (score > 0.1f);
  float ns = 1.f / (1.f + expf(-score));    // re-sigmoid, per reference
  m = m && (ns > 0.1f);
  maskv[id] = m ? 1 : 0;
  clsv[id]  = cls;
  float x1 = xw - d0 * 0.5f, y1 = yw - d1 * 0.5f;
  float* bv = bev4 + id * 4;
  bv[0] = x1; bv[1] = y1; bv[2] = x1 + d0; bv[3] = y1 + d1;
}

// ---------------- greedy BEV NMS per (batch, class) ----------------
__global__ __launch_bounds__(256)
void nms_kernel(const float* __restrict__ bev4, const int* __restrict__ maskv,
                const int* __restrict__ clsv, int* __restrict__ keepAny) {
  int b = blockIdx.x / NCLS, c = blockIdx.x % NCLS;
  __shared__ int   keep[K_];
  __shared__ float bx[K_ * 4];
  int tid = threadIdx.x;
  for (int k = tid; k < K_; k += 256) {
    keep[k] = maskv[b * K_ + k] && (clsv[b * K_ + k] == c);
    bx[k * 4 + 0] = bev4[(b * K_ + k) * 4 + 0];
    bx[k * 4 + 1] = bev4[(b * K_ + k) * 4 + 1];
    bx[k * 4 + 2] = bev4[(b * K_ + k) * 4 + 2];
    bx[k * 4 + 3] = bev4[(b * K_ + k) * 4 + 3];
  }
  __syncthreads();
  for (int i = 0; i < K_; ++i) {
    int ki = keep[i];                 // this iteration only writes k > i, so safe to read
    if (ki) {
      float ax1 = bx[i * 4 + 0], ay1 = bx[i * 4 + 1], ax2 = bx[i * 4 + 2], ay2 = bx[i * 4 + 3];
      float aarea = (ax2 - ax1) * (ay2 - ay1);
      for (int k = i + 1 + tid; k < K_; k += 256) {
        float ix1 = fmaxf(ax1, bx[k * 4 + 0]);
        float iy1 = fmaxf(ay1, bx[k * 4 + 1]);
        float ix2 = fminf(ax2, bx[k * 4 + 2]);
        float iy2 = fminf(ay2, bx[k * 4 + 3]);
        float inter = fmaxf(ix2 - ix1, 0.f) * fmaxf(iy2 - iy1, 0.f);
        float barea = (bx[k * 4 + 2] - bx[k * 4 + 0]) * (bx[k * 4 + 3] - bx[k * 4 + 1]);
        float iou = inter / (aarea + barea - inter + 1e-6f);
        if (iou > 0.1f) keep[k] = 0;
      }
    }
    __syncthreads();
  }
  for (int k = tid; k < K_; k += 256)
    if (keep[k]) atomicOr(&keepAny[b * K_ + k], 1);
}

// ---------------- write padded outputs: rois | scores | labels ----------------
__global__ void finalize_kernel(const float* __restrict__ boxes, const float* __restrict__ tscore,
                                const int* __restrict__ clsv, const int* __restrict__ keepAny,
                                float* __restrict__ out) {
  int id = blockIdx.x * 256 + threadIdx.x;
  if (id >= B_ * K_) return;
  int kp = keepAny[id];
  #pragma unroll
  for (int j = 0; j < 7; ++j)
    out[id * 7 + j] = kp ? boxes[id * 7 + j] : 0.f;
  out[B_ * K_ * 7 + id]           = kp ? tscore[id] : 0.f;
  out[B_ * K_ * 7 + B_ * K_ + id] = kp ? (float)clsv[id] : 0.f;
}

// ---------------- host launcher ----------------
extern "C" void kernel_launch(void* const* d_in, const int* in_sizes, int n_in,
                              void* d_out, int out_size, void* d_ws, size_t ws_size,
                              hipStream_t stream) {
  (void)out_size; (void)ws_size;
  const float* bev = (const float*)d_in[0];
  const float *sw, *sb, *sg, *sbe, *sm, *sv;
  const float *hw0[5], *hb0[5], *hg0[5], *hbe0[5], *hm0[5], *hv0[5], *hwf[5], *hbf[5];

  bool insertion = (n_in > 1 && in_sizes[1] == 64 * 384 * 9);
  if (insertion) {
    // bev, shared{w,b,g,be,m,v}, heads{w0,b0,g0,be0,m0,v0,wf,bf}
    sw  = (const float*)d_in[1]; sb  = (const float*)d_in[2];
    sg  = (const float*)d_in[3]; sbe = (const float*)d_in[4];
    sm  = (const float*)d_in[5]; sv  = (const float*)d_in[6];
    for (int h = 0; h < 5; ++h) {
      int base = 7 + 8 * h;
      hw0[h]  = (const float*)d_in[base + 0]; hb0[h]  = (const float*)d_in[base + 1];
      hg0[h]  = (const float*)d_in[base + 2]; hbe0[h] = (const float*)d_in[base + 3];
      hm0[h]  = (const float*)d_in[base + 4]; hv0[h]  = (const float*)d_in[base + 5];
      hwf[h]  = (const float*)d_in[base + 6]; hbf[h]  = (const float*)d_in[base + 7];
    }
  } else {
    // sorted tree-flatten: bev, heads{b0,be0,bf,g0,m0,v0,w0,wf}x5, shared{b,be,g,m,v,w}
    for (int h = 0; h < 5; ++h) {
      int base = 1 + 8 * h;
      hb0[h]  = (const float*)d_in[base + 0]; hbe0[h] = (const float*)d_in[base + 1];
      hbf[h]  = (const float*)d_in[base + 2]; hg0[h]  = (const float*)d_in[base + 3];
      hm0[h]  = (const float*)d_in[base + 4]; hv0[h]  = (const float*)d_in[base + 5];
      hw0[h]  = (const float*)d_in[base + 6]; hwf[h]  = (const float*)d_in[base + 7];
    }
    sb = (const float*)d_in[41]; sbe = (const float*)d_in[42];
    sg = (const float*)d_in[43]; sm  = (const float*)d_in[44];
    sv = (const float*)d_in[45]; sw  = (const float*)d_in[46];
  }

  // workspace carve-up (256B aligned)
  char* wsb = (char*)d_ws;
  size_t off = 0;
  auto carve = [&](size_t bytes) -> void* {
    void* p = wsb + off;
    off = (off + bytes + 255) & ~(size_t)255;
    return p;
  };
  float*          coefs = (float*)carve(6 * 128 * sizeof(float));
  unsigned short* xbuf  = (unsigned short*)carve((size_t)B_ * P_ * 64 * 2);
  unsigned short* hbuf  = (unsigned short*)carve((size_t)B_ * P_ * 64 * 2);
  float*          preds = (float*)carve((size_t)B_ * NCH * P_ * sizeof(float));
  float*          tsc   = (float*)carve(B_ * K_ * sizeof(float));
  int*            tin   = (int*)carve(B_ * K_ * sizeof(int));
  float*          boxes = (float*)carve(B_ * K_ * 7 * sizeof(float));
  float*          bev4  = (float*)carve(B_ * K_ * 4 * sizeof(float));
  int*            maskv = (int*)carve(B_ * K_ * sizeof(int));
  int*            clsv  = (int*)carve(B_ * K_ * sizeof(int));
  int*            keepA = (int*)carve(B_ * K_ * sizeof(int));

  // 1) fold BN + conv bias into per-channel scale/shift (layer 0 = shared, 1..5 = heads)
  bn_coef_kernel<<<1, 64, 0, stream>>>(sg, sbe, sm, sv, sb, coefs);
  for (int h = 0; h < 5; ++h)
    bn_coef_kernel<<<1, 64, 0, stream>>>(hg0[h], hbe0[h], hm0[h], hv0[h], hb0[h],
                                         coefs + (1 + h) * 128);

  const int tiles = (P_ + 127) / 128;   // 277

  // 2) shared conv 384->64 (NCHW f32 in, NHWC bf16 out) — WMMA implicit GEMM
  conv3x3_bn_relu_wmma<IC0, true><<<dim3(tiles, B_), 256, 0, stream>>>(bev, sw, coefs, xbuf);

  // 3) heads: conv0 64->64 (WMMA + async global->LDS) + final conv 64->oc (direct)
  const int HEAD_OC[5] = {2, 1, 3, 2, 3};
  const int CHB[5]     = {0, 2, 3, 6, 8};
  for (int h = 0; h < 5; ++h) {
    conv3x3_bn_relu_wmma<64, false><<<dim3(tiles, B_), 256, 0, stream>>>(
        xbuf, hw0[h], coefs + (1 + h) * 128, hbuf);
    int total = B_ * P_ * HEAD_OC[h];
    head_final_conv<<<(total + 255) / 256, 256, 0, stream>>>(
        hbuf, hwf[h], hbf[h], preds, HEAD_OC[h], CHB[h]);
  }

  // 4) sigmoid heatmap, 5) exact top-500/batch, 6) decode, 7) NMS, 8) outputs
  int stot = B_ * NCLS * P_;
  hm_sigmoid<<<(stot + 255) / 256, 256, 0, stream>>>(preds);
  topk_kernel<<<B_, 1024, 0, stream>>>(preds, tsc, tin);
  decode_kernel<<<(B_ * K_ + 255) / 256, 256, 0, stream>>>(preds, tsc, tin,
                                                           boxes, bev4, maskv, clsv, keepA);
  nms_kernel<<<B_ * NCLS, 256, 0, stream>>>(bev4, maskv, clsv, keepA);
  finalize_kernel<<<(B_ * K_ + 255) / 256, 256, 0, stream>>>(boxes, tsc, clsv, keepA,
                                                             (float*)d_out);
}
